// GlobalTransformerBlock_75514114998347
// MI455X (gfx1250) — matrix operand order repaired and checked by reading, not compiled, not run on hardware
//
#include <hip/hip_runtime.h>

// ---------------------------------------------------------------------------
// Transformer block for MI455X (gfx1250): bf16 WMMA (16x16x32, f32 accum),
// async-to-LDS cooperative weight staging in the block-uniform GEMMs.
// B=4 S=2048 E=512 H=8 D=64 FFN=2304.
// ---------------------------------------------------------------------------

typedef __attribute__((ext_vector_type(16))) __bf16 v16bf;
typedef __attribute__((ext_vector_type(8)))  __bf16 v8bf;
typedef __attribute__((ext_vector_type(8)))  float  v8f;

#define B_  4
#define S_  2048
#define E_  512
#define H_  8
#define D_  64
#define F_  2304
#define M_  (B_ * S_)          // 8192 token rows

// ---- scalar bf16 <-> f32 helpers (RNE) ------------------------------------
static __device__ __forceinline__ __bf16 f2bf(float f) {
  unsigned u = __builtin_bit_cast(unsigned, f);
  u = u + 0x7fffu + ((u >> 16) & 1u);
  unsigned short h = (unsigned short)(u >> 16);
  return __builtin_bit_cast(__bf16, h);
}
static __device__ __forceinline__ float bf2f(__bf16 b) {
  unsigned short h = __builtin_bit_cast(unsigned short, b);
  unsigned u = ((unsigned)h) << 16;
  return __builtin_bit_cast(float, u);
}

// ---- WMMA wrapper ----------------------------------------------------------
static __device__ __forceinline__ v8f wmma_bf16(v16bf a, v16bf b, v8f c) {
  return __builtin_amdgcn_wmma_f32_16x16x32_bf16(
      /*neg_a=*/false, a, /*neg_b=*/false, b,
      /*c_mod=*/(short)0, c, /*reuse_a=*/false, /*reuse_b=*/false);
}

// ---- CDNA5 async copy: global -> LDS, 16B per lane, tracked by ASYNCcnt ----
static __device__ __forceinline__ void async_load_b128(unsigned lds_off,
                                                       const void* gptr) {
  asm volatile("global_load_async_to_lds_b128 %0, %1, off"
               :: "v"(lds_off),
                  "v"((unsigned long long)(__SIZE_TYPE__)gptr)
               : "memory");
}
static __device__ __forceinline__ void wait_asynccnt0() {
  asm volatile("s_wait_asynccnt 0x0" ::: "memory");
}

// A fragment: 16x32 bf16, row-major source, row m = lane&15.
// lanes 0-15 hold K 0..15 contiguous, lanes 16-31 K 16..31.
static __device__ __forceinline__ v16bf frag_a(const __bf16* __restrict__ A,
                                               int lda, int m0, int k0, int lane) {
  int m  = m0 + (lane & 15);
  int kb = k0 + ((lane >> 4) << 4);
  const __bf16* p = A + m * lda + kb;
  v16bf r;
  ((v8bf*)&r)[0] = *(const v8bf*)(p);
  ((v8bf*)&r)[1] = *(const v8bf*)(p + 8);
  return r;
}
// B fragment: 32x16 bf16 (K x N). Column n = lane&15 taken from row n of W [NxK]
// (i.e. computes A @ W^T). lanes 0-15 hold K 0..15 contiguous, lanes 16-31 K 16..31.
static __device__ __forceinline__ v16bf frag_b(const __bf16* __restrict__ W,
                                               int ldw, int n0, int k0, int lane) {
  int n  = n0 + (lane & 15);
  int kb = k0 + ((lane >> 4) << 4);
  const __bf16* p = W + n * ldw + kb;
  v16bf r;
  ((v8bf*)&r)[0] = *(const v8bf*)(p);
  ((v8bf*)&r)[1] = *(const v8bf*)(p + 8);
  return r;
}
// Same as frag_b but from an LDS-staged tile with row stride 32 halves.
static __device__ __forceinline__ v16bf frag_b_lds(const __bf16* base,
                                                   int n0l, int lane) {
  int n  = n0l + (lane & 15);
  int kb = (lane >> 4) << 4;
  const __bf16* p = base + n * 32 + kb;
  v16bf r;
  ((v8bf*)&r)[0] = *(const v8bf*)(p);
  ((v8bf*)&r)[1] = *(const v8bf*)(p + 8);
  return r;
}

// ---------------------------------------------------------------------------
// f32 -> bf16 convert (weights)
// ---------------------------------------------------------------------------
__global__ void __launch_bounds__(256)
k_f32_to_bf16(const float* __restrict__ in, __bf16* __restrict__ out, int n) {
  int i = blockIdx.x * 256 + threadIdx.x;
  if (i < n) out[i] = f2bf(in[i]);
}

// ---------------------------------------------------------------------------
// RMSNorm: one wave per 512-wide row, output bf16
// ---------------------------------------------------------------------------
__global__ void __launch_bounds__(128)
k_rmsnorm(const float* __restrict__ x, const float* __restrict__ w,
          __bf16* __restrict__ out) {
  int lane = threadIdx.x & 31;
  int row  = blockIdx.x * 4 + (threadIdx.x >> 5);
  const float4* xr = (const float4*)(x + row * E_);
  const float4* wr = (const float4*)(w);
  float4 v[4];
  float ss = 0.f;
#pragma unroll
  for (int i = 0; i < 4; ++i) {
    v[i] = xr[lane + 32 * i];
    ss += v[i].x * v[i].x + v[i].y * v[i].y + v[i].z * v[i].z + v[i].w * v[i].w;
  }
#pragma unroll
  for (int off = 16; off >= 1; off >>= 1) ss += __shfl_xor(ss, off, 32);
  float r = rsqrtf(ss * (1.0f / E_) + 1e-6f);
#pragma unroll
  for (int i = 0; i < 4; ++i) {
    float4 wv = wr[lane + 32 * i];
    __bf16* o = out + row * E_ + (lane + 32 * i) * 4;
    o[0] = f2bf(v[i].x * r * wv.x);
    o[1] = f2bf(v[i].y * r * wv.y);
    o[2] = f2bf(v[i].z * r * wv.z);
    o[3] = f2bf(v[i].w * r * wv.w);
  }
}

// ---------------------------------------------------------------------------
// QKV GEMM: qkv = h @ Wqkv^T. Epilogue scatters to
//   qk_raw [2][B][H][S][64] (q,k, pre-RoPE) and vt [B][H][64][S] (v transposed)
// Wave tile: 32(M) x 64(N). Block = 4 waves stacked in M.
// ---------------------------------------------------------------------------
__global__ void __launch_bounds__(128)
k_gemm_qkv(const __bf16* __restrict__ A, const __bf16* __restrict__ W,
           __bf16* __restrict__ qk_raw, __bf16* __restrict__ vt) {
  int lane = threadIdx.x & 31;
  int wave = threadIdx.x >> 5;
  int m0 = (blockIdx.x * 4 + wave) * 32;
  int n0 = blockIdx.y * 64;
  v8f acc[2][4] = {};
  for (int k0 = 0; k0 < E_; k0 += 32) {
    v16bf a0 = frag_a(A, E_, m0,      k0, lane);
    v16bf a1 = frag_a(A, E_, m0 + 16, k0, lane);
#pragma unroll
    for (int j = 0; j < 4; ++j) {
      v16bf b = frag_b(W, E_, n0 + 16 * j, k0, lane);
      acc[0][j] = wmma_bf16(a0, b, acc[0][j]);
      acc[1][j] = wmma_bf16(a1, b, acc[1][j]);
    }
  }
  int hi = lane >> 4, nl = lane & 15;
#pragma unroll
  for (int t = 0; t < 2; ++t)
#pragma unroll
    for (int j = 0; j < 4; ++j) {
      int n = n0 + 16 * j + nl;
      int which = n >> 9;          // 0=q 1=k 2=v
      int e = n & 511;
      int h = e >> 6, d = e & 63;
#pragma unroll
      for (int r = 0; r < 8; ++r) {
        int m = m0 + 16 * t + r + 8 * hi;
        int b = m >> 11, s = m & (S_ - 1);
        float val = acc[t][j][r];
        if (which == 2) {
          vt[(((b * H_ + h) * D_ + d) << 11) + s] = f2bf(val);
        } else {
          qk_raw[((((which * B_ + b) * H_ + h) << 11) + s) * D_ + d] = f2bf(val);
        }
      }
    }
}

// ---------------------------------------------------------------------------
// RoPE on q,k (interleaved-pair rotate_half semantics)
// ---------------------------------------------------------------------------
__global__ void __launch_bounds__(256)
k_rope(const __bf16* __restrict__ raw, __bf16* __restrict__ roped) {
  int idx  = blockIdx.x * 256 + threadIdx.x;   // 2*B*H*S*32 threads
  int i    = idx & 31;
  int s    = (idx >> 5) & (S_ - 1);
  int rest = idx >> 16;                        // [2*B*H)
  int base = (rest << 17) + (s << 6);
  int d0 = 2 * i, d1 = d0 + 1;
  float x0 = bf2f(raw[base + d0]);
  float x1 = bf2f(raw[base + d1]);
  const float c = 0.28782313662425575f;        // ln(10000)/32
  float th0 = (float)s * __expf(-c * (float)(d0 & 31));
  float th1 = (float)s * __expf(-c * (float)(d1 & 31));
  roped[base + d0] = f2bf(x0 * cosf(th0) - x1 * sinf(th0));
  roped[base + d1] = f2bf(x1 * cosf(th1) + x0 * sinf(th1));
}

// ---------------------------------------------------------------------------
// Causal flash attention. One wave owns 16 q-rows of one (b,h).
// S = Q K^T via WMMA (d=64 -> 2 K-steps); online softmax in f32; P tile goes
// C-layout -> LDS -> A-layout (bf16); ctx += P @ V via WMMA against vt [64][S].
// Per-wave LDS protocol (no block barrier: waves have divergent trip counts).
// ---------------------------------------------------------------------------
__global__ void __launch_bounds__(128)
k_attn(const __bf16* __restrict__ qr, const __bf16* __restrict__ kr,
       const __bf16* __restrict__ vt, __bf16* __restrict__ ctx) {
  __shared__ __align__(16) __bf16 lds_p[4][16 * 32];
  int lane = threadIdx.x & 31;
  int wave = threadIdx.x >> 5;
  int gw   = blockIdx.x * 4 + wave;            // [0, B*H*S/16)
  int m0   = (gw & ((S_ / 16) - 1)) << 4;
  int bh   = gw >> 7;
  const __bf16* Q  = qr + ((long)bh << 17);    // [S][64]
  const __bf16* Km = kr + ((long)bh << 17);    // [S][64]
  const __bf16* Vt = vt + ((long)bh << 17);    // [64][S]
  __bf16* pw = &lds_p[wave][0];

  v16bf qf0 = frag_a(Q, D_, m0, 0,  lane);
  v16bf qf1 = frag_a(Q, D_, m0, 32, lane);
  v8f cacc[4] = {};
  float mrun[8], lrun[8];
#pragma unroll
  for (int r = 0; r < 8; ++r) { mrun[r] = -1e30f; lrun[r] = 0.f; }

  int hi = lane >> 4, nl = lane & 15;
  int nkb = ((m0 + 15) >> 5) + 1;              // causal 32-wide kpos blocks
  for (int kb = 0; kb < nkb; ++kb) {
    int kpos0 = kb * 32;
    v8f sacc[2] = {};
#pragma unroll
    for (int u = 0; u < 2; ++u) {
      v16bf b0 = frag_b(Km, D_, kpos0 + 16 * u, 0,  lane);
      v16bf b1 = frag_b(Km, D_, kpos0 + 16 * u, 32, lane);
      sacc[u] = wmma_bf16(qf0, b0, sacc[u]);
      sacc[u] = wmma_bf16(qf1, b1, sacc[u]);
    }
    // online softmax update, row by row (row m lives on a 16-lane half)
#pragma unroll
    for (int r = 0; r < 8; ++r) {
      int m = m0 + r + 8 * hi;
      float s0 = sacc[0][r] * 0.125f;
      float s1 = sacc[1][r] * 0.125f;
      if (kpos0 + nl      > m) s0 = -1e30f;
      if (kpos0 + 16 + nl > m) s1 = -1e30f;
      float mx = fmaxf(s0, s1);
#pragma unroll
      for (int off = 1; off < 16; off <<= 1) mx = fmaxf(mx, __shfl_xor(mx, off, 32));
      float newm = fmaxf(mrun[r], mx);
      float corr = __expf(mrun[r] - newm);
      float p0 = __expf(s0 - newm);
      float p1 = __expf(s1 - newm);
      float rs = p0 + p1;
#pragma unroll
      for (int off = 1; off < 16; off <<= 1) rs += __shfl_xor(rs, off, 32);
      lrun[r] = lrun[r] * corr + rs;
      mrun[r] = newm;
#pragma unroll
      for (int dt = 0; dt < 4; ++dt) cacc[dt][r] *= corr;
      pw[(r + 8 * hi) * 32 + nl]      = f2bf(p0);
      pw[(r + 8 * hi) * 32 + 16 + nl] = f2bf(p1);
    }
    // C-layout -> A-layout through LDS (DS ops are in-order per wave;
    // barriers stop compiler reordering of the cross-lane traffic)
    __asm__ volatile("" ::: "memory");
    __builtin_amdgcn_wave_barrier();
    v16bf pf;
    {
      int mm = lane & 15;
      int kbse = (lane >> 4) << 4;
      const __bf16* p = pw + mm * 32 + kbse;
      ((v8bf*)&pf)[0] = *(const v8bf*)(p);
      ((v8bf*)&pf)[1] = *(const v8bf*)(p + 8);
    }
#pragma unroll
    for (int dt = 0; dt < 4; ++dt) {
      int n = dt * 16 + nl;
      int kbse = (lane >> 4) << 4;
      const __bf16* p = Vt + (long)n * S_ + kpos0 + kbse;
      v16bf bv;
      ((v8bf*)&bv)[0] = *(const v8bf*)(p);
      ((v8bf*)&bv)[1] = *(const v8bf*)(p + 8);
      cacc[dt] = wmma_bf16(pf, bv, cacc[dt]);
    }
    __asm__ volatile("" ::: "memory");
    __builtin_amdgcn_wave_barrier();
  }
  // epilogue: ctx[b, s, h*64 + d], normalized by row sum
  int b = bh >> 3, h = bh & 7;
#pragma unroll
  for (int r = 0; r < 8; ++r) {
    int s = m0 + r + 8 * hi;
    float inv = 1.f / lrun[r];
#pragma unroll
    for (int dt = 0; dt < 4; ++dt) {
      ctx[(b * S_ + s) * E_ + h * D_ + dt * 16 + nl] = f2bf(cacc[dt][r] * inv);
    }
  }
}

// ---------------------------------------------------------------------------
// GEMM + f32 residual: out[M,512] = resid + A_bf[M,K] @ W_bf[512,K]^T
// B tile (64N x 32K, shared by all 4 waves) staged via async-to-LDS, double
// buffered; loads of tile k+1 overlap WMMA on tile k.
// ---------------------------------------------------------------------------
__global__ void __launch_bounds__(128)
k_gemm_resid(const __bf16* __restrict__ A, const __bf16* __restrict__ W,
             const float* __restrict__ resid, float* __restrict__ out, int K) {
  __shared__ __align__(16) __bf16 lds_b[2][64 * 32];
  int lane = threadIdx.x & 31;
  int wave = threadIdx.x >> 5;
  int tid  = threadIdx.x;
  int m0 = (blockIdx.x * 4 + wave) * 32;
  int n0 = blockIdx.y * 64;

  // 64 rows x 64B = 256 x 16B lane-transfers -> 2 per thread per k-step
  auto stage = [&](int buf, int k0) {
#pragma unroll
    for (int q = 0; q < 2; ++q) {
      int idx = tid + 128 * q;
      int row = idx >> 2, seg = (idx & 3) * 8;
      async_load_b128((unsigned)(__SIZE_TYPE__)&lds_b[buf][row * 32 + seg],
                      W + (n0 + row) * K + k0 + seg);
    }
  };

  int nk = K / 32;
  stage(0, 0);
  v8f acc[2][4] = {};
  for (int ks = 0; ks < nk; ++ks) {
    wait_asynccnt0();          // own share of tile ks landed
    __syncthreads();           // everyone's share landed; prev consume done
    if (ks + 1 < nk) stage((ks + 1) & 1, (ks + 1) * 32);
    int k0 = ks * 32;
    v16bf a0 = frag_a(A, K, m0,      k0, lane);
    v16bf a1 = frag_a(A, K, m0 + 16, k0, lane);
    const __bf16* bb = &lds_b[ks & 1][0];
#pragma unroll
    for (int j = 0; j < 4; ++j) {
      v16bf b = frag_b_lds(bb, 16 * j, lane);
      acc[0][j] = wmma_bf16(a0, b, acc[0][j]);
      acc[1][j] = wmma_bf16(a1, b, acc[1][j]);
    }
  }
  int hi = lane >> 4, nl = lane & 15;
#pragma unroll
  for (int t = 0; t < 2; ++t)
#pragma unroll
    for (int j = 0; j < 4; ++j) {
      int n = n0 + 16 * j + nl;
#pragma unroll
      for (int r = 0; r < 8; ++r) {
        int m = m0 + 16 * t + r + 8 * hi;
        out[m * E_ + n] = resid[m * E_ + n] + acc[t][j][r];
      }
    }
}

// ---------------------------------------------------------------------------
// Fused gate/up GEMM: hidden = silu(h2 @ Wg^T) * (h2 @ Wu^T), bf16 out [M,2304]
// Wave tile 32(M) x 32(N); Wg and Wu tiles staged together via async-to-LDS
// (rows 0-31 = Wg, rows 32-63 = Wu), double buffered.
// ---------------------------------------------------------------------------
__global__ void __launch_bounds__(128)
k_gemm_gateup(const __bf16* __restrict__ A, const __bf16* __restrict__ Wg,
              const __bf16* __restrict__ Wu, __bf16* __restrict__ hid) {
  __shared__ __align__(16) __bf16 lds_b[2][64 * 32];
  int lane = threadIdx.x & 31;
  int wave = threadIdx.x >> 5;
  int tid  = threadIdx.x;
  int m0 = (blockIdx.x * 4 + wave) * 32;
  int n0 = blockIdx.y * 32;

  auto stage = [&](int buf, int k0) {
    int row = tid >> 2, seg = (tid & 3) * 8;   // 32 rows x 4 segs = 128 threads
    async_load_b128((unsigned)(__SIZE_TYPE__)&lds_b[buf][row * 32 + seg],
                    Wg + (n0 + row) * E_ + k0 + seg);
    async_load_b128((unsigned)(__SIZE_TYPE__)&lds_b[buf][(32 + row) * 32 + seg],
                    Wu + (n0 + row) * E_ + k0 + seg);
  };

  int nk = E_ / 32;
  stage(0, 0);
  v8f ag[2][2] = {}, au[2][2] = {};
  for (int ks = 0; ks < nk; ++ks) {
    wait_asynccnt0();
    __syncthreads();
    if (ks + 1 < nk) stage((ks + 1) & 1, (ks + 1) * 32);
    int k0 = ks * 32;
    v16bf a0 = frag_a(A, E_, m0,      k0, lane);
    v16bf a1 = frag_a(A, E_, m0 + 16, k0, lane);
    const __bf16* bb = &lds_b[ks & 1][0];
#pragma unroll
    for (int j = 0; j < 2; ++j) {
      v16bf bg = frag_b_lds(bb,             16 * j, lane);
      v16bf bu = frag_b_lds(bb + 32 * 32,   16 * j, lane);
      ag[0][j] = wmma_bf16(a0, bg, ag[0][j]);
      ag[1][j] = wmma_bf16(a1, bg, ag[1][j]);
      au[0][j] = wmma_bf16(a0, bu, au[0][j]);
      au[1][j] = wmma_bf16(a1, bu, au[1][j]);
    }
  }
  int hi = lane >> 4, nl = lane & 15;
#pragma unroll
  for (int t = 0; t < 2; ++t)
#pragma unroll
    for (int j = 0; j < 2; ++j) {
      int n = n0 + 16 * j + nl;
#pragma unroll
      for (int r = 0; r < 8; ++r) {
        int m = m0 + 16 * t + r + 8 * hi;
        float g = ag[t][j][r];
        float u = au[t][j][r];
        float sg = g / (1.f + __expf(-g));     // silu
        hid[m * F_ + n] = f2bf(sg * u);
      }
    }
}

// ---------------------------------------------------------------------------
// Host launcher
// ---------------------------------------------------------------------------
extern "C" void kernel_launch(void* const* d_in, const int* in_sizes, int n_in,
                              void* d_out, int out_size, void* d_ws, size_t ws_size,
                              hipStream_t stream) {
  (void)in_sizes; (void)n_in; (void)out_size; (void)ws_size;
  const float* x    = (const float*)d_in[0];
  const float* qkvw = (const float*)d_in[1];
  const float* outw = (const float*)d_in[2];
  const float* n1w  = (const float*)d_in[3];
  const float* n2w  = (const float*)d_in[4];
  const float* gw   = (const float*)d_in[5];
  const float* uw   = (const float*)d_in[6];
  const float* dw   = (const float*)d_in[7];
  float* out = (float*)d_out;

  char* p = (char*)d_ws;
  auto carve = [&](size_t nelem, size_t esize) {
    void* r = (void*)p; p += nelem * esize; return r;
  };
  __bf16* WQ  = (__bf16*)carve((size_t)3 * E_ * E_, 2);
  __bf16* WO  = (__bf16*)carve((size_t)E_ * E_, 2);
  __bf16* WG  = (__bf16*)carve((size_t)F_ * E_, 2);
  __bf16* WU  = (__bf16*)carve((size_t)F_ * E_, 2);
  __bf16* WD  = (__bf16*)carve((size_t)E_ * F_, 2);
  __bf16* HB  = (__bf16*)carve((size_t)M_ * E_, 2);      // rmsnorm1(x)
  __bf16* QK  = (__bf16*)carve((size_t)2 * M_ * E_, 2);  // q,k pre-RoPE
  __bf16* QR  = (__bf16*)carve((size_t)2 * M_ * E_, 2);  // q,k post-RoPE
  __bf16* VT  = (__bf16*)carve((size_t)M_ * E_, 2);      // v transposed [B,H,64,S]
  __bf16* CTX = (__bf16*)carve((size_t)M_ * E_, 2);      // attention output
  __bf16* H2  = (__bf16*)carve((size_t)M_ * E_, 2);      // rmsnorm2(x1)
  __bf16* HID = (__bf16*)carve((size_t)M_ * F_, 2);      // silu(gate)*up
  float*  X1  = (float*)carve((size_t)M_ * E_, 4);       // x + attn residual

  auto cvt = [&](const float* s, __bf16* d, int n) {
    k_f32_to_bf16<<<(n + 255) / 256, 256, 0, stream>>>(s, d, n);
  };
  cvt(qkvw, WQ, 3 * E_ * E_);
  cvt(outw, WO, E_ * E_);
  cvt(gw,   WG, F_ * E_);
  cvt(uw,   WU, F_ * E_);
  cvt(dw,   WD, E_ * F_);

  k_rmsnorm<<<M_ / 4, 128, 0, stream>>>(x, n1w, HB);
  k_gemm_qkv<<<dim3(M_ / 128, (3 * E_) / 64), 128, 0, stream>>>(HB, WQ, QK, VT);
  k_rope<<<(2 * M_ * E_ / 2) / 256, 256, 0, stream>>>(QK, QR);
  k_attn<<<(B_ * H_ * (S_ / 16)) / 4, 128, 0, stream>>>(QR, QR + (size_t)M_ * E_, VT, CTX);
  k_gemm_resid<<<dim3(M_ / 128, E_ / 64), 128, 0, stream>>>(CTX, WO, x, X1, E_);
  k_rmsnorm<<<M_ / 4, 128, 0, stream>>>(X1, n2w, H2);
  k_gemm_gateup<<<dim3(M_ / 128, F_ / 32), 128, 0, stream>>>(H2, WG, WU, HID);
  k_gemm_resid<<<dim3(M_ / 128, E_ / 64), 128, 0, stream>>>(HID, WD, X1, out, F_);
}
